// Router_70626442215503
// MI455X (gfx1250) — matrix operand order repaired and checked by reading, not compiled, not run on hardware
//
#include <hip/hip_runtime.h>
#include <hip/hip_bf16.h>
#include <math.h>

// ---------------------------------------------------------------------------
// MoE router for MI455X (gfx1250, wave32, WMMA):
//   logits = x @ w1^T + b_lin   (bf16 WMMA, f32 accum)
//   s = sigmoid(logits); s /= sum(s); s += bias
//   group-limited top-k (8 groups of 32, keep 4), top-8, values = s*2.5
//
// Pipeline (1 barrier / K-chunk, double-buffered LDS stage, K-loop unrolled
// by 2 so the buffer index is a literal and all LDS fragment addresses fold
// into ds_load_b128 immediates -> no address VALU / WAR-hazard NOPs in the
// WMMA chain):
//   fetch(k+1) -> barrier -> WMMA(k, buf A) -> cvt+store(k+1 -> buf B)
// ---------------------------------------------------------------------------

typedef __bf16 v16bf __attribute__((ext_vector_type(16)));
typedef __bf16 v4bf  __attribute__((ext_vector_type(4)));
typedef float  v4f   __attribute__((ext_vector_type(4)));
typedef float  v8f   __attribute__((ext_vector_type(8)));

#define N_TOKENS   32768
#define DIM        2048
#define N_EXPERTS  256
#define N_GROUPS   8
#define GROUP_SIZE 32
#define N_LIM      4
#define TOPK       8
#define ROUTE_SCALE 2.5f

#define TILE_M  64    // tokens per block
#define KC      32    // K chunk (matches 16x16x32 bf16 WMMA)
#define THREADS 256   // 8 waves of 32

#define X_ITERS ((TILE_M * KC) / 4 / THREADS)     // 2 float4 per thread
#define W_ITERS ((N_EXPERTS * KC) / 4 / THREADS)  // 8 float4 per thread

// Double-buffered staging (2 x 20 KB) is dead after the K loop; the 64 KB
// logits tile reuses the same LDS. alignas(64): v16bf loads -> ds_load_b128.
struct Stage {
  __bf16 xs[TILE_M * KC];     // 4 KB,  natural rows, same k order as ws
  __bf16 ws[N_EXPERTS * KC];  // 16 KB, natural rows == B-frag order
};
union alignas(64) Smem {
  Stage stage[2];                   // 40 KB
  float logits[TILE_M * N_EXPERTS]; // 64 KB
};

__global__ __launch_bounds__(THREADS, 1)
void router_gemm_topk(const float* __restrict__ x,
                      const float* __restrict__ w1,
                      const float* __restrict__ b_lin,
                      const float* __restrict__ bias,
                      float* __restrict__ out_vals,
                      int*   __restrict__ out_idx)
{
  __shared__ Smem sm;
  const int tid  = threadIdx.x;
  const int lane = tid & 31;
  const int wave = tid >> 5;
  const int tok0 = blockIdx.x * TILE_M;

  // Wave w -> 16-token row tile (mt) x eight 16-expert column tiles.
  const int mt     = wave >> 1;          // 0..3
  const int ntBase = (wave & 1) * 8;     // 0 or 8 (of 16 column tiles)

  const int half = lane >> 4;            // 0: lanes 0-15, 1: lanes 16-31
  const int l15  = lane & 15;

  // Loop-invariant fragment offsets (elements). nt / buffer deltas are
  // compile-time constants that fold into ds_load immediates.
  const int aOff = (mt * 16 + l15) * KC + half * 16;
  const int bOff = (ntBase * 16 + l15) * KC + half * 16;

  v8f acc[8] = {};                       // 64 VGPRs of f32 accumulators

  float4 xreg[X_ITERS];
  float4 wreg[W_ITERS];

  // ---- global prefetch of one K-chunk into registers (float4 per thread)
  auto fetch = [&](int kb) {
    #pragma unroll
    for (int it = 0; it < X_ITERS; ++it) {
      const int i = tid + it * THREADS;
      const int row = i >> 3, q = i & 7;   // 8 quads per 32-elem row
      xreg[it] = *(const float4*)(x + (size_t)(tok0 + row) * DIM + kb + q * 4);
    }
    #pragma unroll
    for (int it = 0; it < W_ITERS; ++it) {
      const int i = tid + it * THREADS;
      const int row = i >> 3, q = i & 7;
      wreg[it] = *(const float4*)(w1 + (size_t)row * DIM + kb + q * 4);
    }
  };

  // ---- convert & store staged registers into LDS buffer `buf`.
  //      A-frag row order (bytes): [K0-15] | [K16-31] (natural, matches B)
  auto storeStage = [&](int buf) {
    Stage& st = sm.stage[buf];
    #pragma unroll
    for (int it = 0; it < X_ITERS; ++it) {
      const int i = tid + it * THREADS;
      const int row = i >> 3, q = i & 7;
      const int b  = q >> 1;                             // 8-elem block 0..3
      const int pb = b;
      const v4f f = { xreg[it].x, xreg[it].y, xreg[it].z, xreg[it].w };
      *(v4bf*)&st.xs[row * KC + (pb * 2 + (q & 1)) * 4] =
          __builtin_convertvector(f, v4bf);
    }
    #pragma unroll
    for (int it = 0; it < W_ITERS; ++it) {
      const int i = tid + it * THREADS;
      const int row = i >> 3, q = i & 7;
      const v4f f = { wreg[it].x, wreg[it].y, wreg[it].z, wreg[it].w };
      *(v4bf*)&st.ws[row * KC + q * 4] =
          __builtin_convertvector(f, v4bf);
    }
  };

// One pipeline step with a LITERAL buffer index: fetch k+1, barrier, 8 WMMAs
// from stage[BUF] (addresses = base VGPR + immediate), store k+1 to the
// other buffer while the XDL pipe drains.
#define STEP(KB, BUF)                                                        \
  {                                                                          \
    const bool more = (KB) + KC < DIM;                                       \
    if (more) fetch((KB) + KC);    /* HBM loads fly during the WMMAs */      \
    __syncthreads();               /* stage[BUF] visible; other buf free */  \
    {                                                                        \
      const Stage& st = sm.stage[(BUF)];                                     \
      const v16bf a = *(const v16bf*)(&st.xs[aOff]);                         \
      v16bf bfr[8];                                                          \
      _Pragma("unroll")                                                      \
      for (int nt = 0; nt < 8; ++nt)                                         \
        bfr[nt] = *(const v16bf*)(&st.ws[bOff + nt * 16 * KC]);              \
      _Pragma("unroll")                                                      \
      for (int nt = 0; nt < 8; ++nt)                                         \
        acc[nt] = __builtin_amdgcn_wmma_f32_16x16x32_bf16(                   \
            false, a, false, bfr[nt], (short)0, acc[nt], false, false);      \
    }                                                                        \
    if (more) storeStage((BUF) ^ 1);                                         \
  }

  // ---- prologue: chunk 0 into buffer 0
  fetch(0);
  storeStage(0);

  // DIM/KC = 64 chunks (even) -> unroll by 2 so BUF is a constant.
  for (int kb = 0; kb < DIM; kb += 2 * KC) {
    STEP(kb, 0);
    STEP(kb + KC, 1);
  }
#undef STEP

  __syncthreads();
  // ---- scatter accumulators to the logits tile.
  //      C layout: lanes 0-15 hold M=r, lanes 16-31 hold M=8+r; N = lane&15.
  #pragma unroll
  for (int nt = 0; nt < 8; ++nt) {
    const int col = (ntBase + nt) * 16 + l15;
    #pragma unroll
    for (int r = 0; r < 8; ++r) {
      const int rowl = mt * 16 + r + half * 8;
      sm.logits[rowl * N_EXPERTS + col] = acc[nt][r];
    }
  }
  __syncthreads();

  // ---- routing epilogue: one thread per token (cheap vs the GEMM).
  if (tid < TILE_M) {
    float* srow = &sm.logits[tid * N_EXPERTS];

    // sigmoid(logit + b_lin), accumulate L1 sum
    float sum = 0.f;
    for (int e = 0; e < N_EXPERTS; ++e) {
      const float s = 1.f / (1.f + __expf(-(srow[e] + b_lin[e])));
      srow[e] = s;
      sum += s;
    }
    const float inv = 1.f / sum;

    // normalize + bias, track per-group max
    float gmax[N_GROUPS];
    for (int g = 0; g < N_GROUPS; ++g) {
      float m = -INFINITY;
      for (int j = 0; j < GROUP_SIZE; ++j) {
        const int e = g * GROUP_SIZE + j;
        const float s = srow[e] * inv + bias[e];
        srow[e] = s;
        m = fmaxf(m, s);
      }
      gmax[g] = m;
    }

    // keep the best N_LIM groups (strict '>' => lowest index wins ties,
    // matching jax.lax.top_k)
    unsigned keep = 0;
    for (int t = 0; t < N_LIM; ++t) {
      float best = -INFINITY; int bg = 0;
      for (int g = 0; g < N_GROUPS; ++g)
        if (!((keep >> g) & 1) && gmax[g] > best) { best = gmax[g]; bg = g; }
      keep |= 1u << bg;
    }

    // destructive top-8 over kept groups (ascending-index scan)
    const long row_g = tok0 + tid;
    for (int t = 0; t < TOPK; ++t) {
      float best = -INFINITY; int bi = 0;
      for (int g = 0; g < N_GROUPS; ++g) {
        if (!((keep >> g) & 1)) continue;
        for (int j = 0; j < GROUP_SIZE; ++j) {
          const int e = g * GROUP_SIZE + j;
          const float v = srow[e];
          if (v > best) { best = v; bi = e; }
        }
      }
      out_vals[row_g * TOPK + t] = best * ROUTE_SCALE;
      out_idx [row_g * TOPK + t] = bi;
      srow[bi] = -INFINITY;
    }
  }
}

extern "C" void kernel_launch(void* const* d_in, const int* in_sizes, int n_in,
                              void* d_out, int out_size, void* d_ws, size_t ws_size,
                              hipStream_t stream) {
  const float* x     = (const float*)d_in[0];
  const float* w1    = (const float*)d_in[1];
  const float* b_lin = (const float*)d_in[2];
  const float* bias  = (const float*)d_in[3];

  float* out_vals = (float*)d_out;                                   // [N, 8] f32
  int*   out_idx  = (int*)((float*)d_out + (size_t)N_TOKENS * TOPK); // [N, 8] i32

  dim3 grid(N_TOKENS / TILE_M);   // 512 blocks
  dim3 block(THREADS);            // 8 waves
  hipLaunchKernelGGL(router_gemm_topk, grid, block, 0, stream,
                     x, w1, b_lin, bias, out_vals, out_idx);
}
